// ModelSingle_76605036691745
// MI455X (gfx1250) — matrix operand order repaired and checked by reading, not compiled, not run on hardware
//
#include <hip/hip_runtime.h>
#include <math.h>

#define B_   64
#define S_   512
#define H_   512
#define T_   (B_*S_)      // 32768 tokens
#define DM_  128
#define DA_  256
#define DE_  256
#define G4_  (4*H_)       // 2048
#define NA_  16
#define NM_  9
#define NT_  32
#define NBINS_ 50

typedef __bf16 bf16_t;
typedef __attribute__((ext_vector_type(16))) __bf16 v16bf;
typedef __attribute__((ext_vector_type(8)))  __bf16 v8bf;
typedef __attribute__((ext_vector_type(8)))  float  v8f;

__device__ __forceinline__ float sigmoidf_(float x){ return 1.0f/(1.0f+__expf(-x)); }

__device__ __forceinline__ float wredf(float v){
  #pragma unroll
  for (int m = 16; m >= 1; m >>= 1) v += __shfl_xor(v, m, 32);
  return v;
}

// ---------------- elementwise / prep kernels ----------------

__global__ void k_symlog_bf16(const float* __restrict__ x, bf16_t* __restrict__ y, int n){
  int i = blockIdx.x*blockDim.x + threadIdx.x;
  if (i >= n) return;
  float v = x[i];
  y[i] = (bf16_t)copysignf(log1pf(fabsf(v)), v);
}

// W [K x N] fp32 row-major  ->  Wt [N x K] bf16 row-major
__global__ void k_transpose_bf16(const float* __restrict__ W, bf16_t* __restrict__ Wt, int K, int N){
  int i = blockIdx.x*blockDim.x + threadIdx.x;
  if (i >= K*N) return;
  int nn = i / K, kk = i % K;
  Wt[i] = (bf16_t)W[(size_t)kk*N + nn];
}

__global__ void k_transpose_f32(const float* __restrict__ W, float* __restrict__ Wt, int K, int N){
  int i = blockIdx.x*blockDim.x + threadIdx.x;
  if (i >= K*N) return;
  int nn = i / K, kk = i % K;
  Wt[i] = W[(size_t)kk*N + nn];
}

__global__ void k_addbias(const float* __restrict__ a, const float* __restrict__ b,
                          float* __restrict__ o, int n){
  int i = blockIdx.x*blockDim.x + threadIdx.x;
  if (i < n) o[i] = a[i] + b[i];
}

__global__ void k_init_state(const float* __restrict__ hx, const float* __restrict__ cx,
                             bf16_t* __restrict__ hbf, float* __restrict__ cst, int n){
  int i = blockIdx.x*blockDim.x + threadIdx.x;
  if (i >= n) return;
  hbf[i] = (bf16_t)hx[i];
  cst[i] = cx[i];
}

// ---------------- bf16 WMMA GEMM: C = A[MxK] @ Bt^T (+bias) ----------------
// A row-major [M x K] bf16 ; Bt row-major [N x K] bf16 (pre-transposed weight).
// Each wave: one 16-row M tile x four 16-col N tiles; A fragment reused 4x.
// N must be a multiple of 64.
__global__ void k_gemm_bf16(const bf16_t* __restrict__ A, const bf16_t* __restrict__ Bt,
                            const float* __restrict__ bias,
                            float* __restrict__ Cf, bf16_t* __restrict__ Cbf,
                            int M, int N, int K, int ldc, int coloff){
  int wid = blockIdx.x*(blockDim.x >> 5) + (threadIdx.x >> 5);
  int ng = N >> 6;                       // groups of 4 n-tiles (64 cols)
  if (wid >= (M >> 4)*ng) return;
  int tm = wid / ng, tg = wid % ng;
  int lane = threadIdx.x & 31;
  int lrow = lane & 15;
  int kA = (lane >> 4) * 16;             // A: lanes 16-31 hold K=16..31
  int kB = (lane >> 4) * 16;             // B: lanes 16-31 hold K=16..31
  const bf16_t* pa = A  + (size_t)(tm*16 + lrow)*K + kA;
  const bf16_t* pb = Bt + (size_t)(tg*64 + lrow)*K + kB;
  v8f acc[4] = {};
  for (int k0 = 0; k0 < K; k0 += 32){
    __builtin_prefetch(pa + k0 + 128, 0, 1);
    v8bf alo = *(const v8bf*)(pa + k0);        // K = k0+kA+0..7
    v8bf ahi = *(const v8bf*)(pa + k0 + 8);    // K = k0+kA+8..15
    v16bf afr;
    #pragma unroll
    for (int i = 0; i < 8; i++){ afr[i] = alo[i]; afr[i+8] = ahi[i]; }
    #pragma unroll
    for (int j = 0; j < 4; j++){
      v16bf bfr = *(const v16bf*)(pb + (size_t)j*16*K + k0);  // K = k0+kB+0..15
      acc[j] = __builtin_amdgcn_wmma_f32_16x16x32_bf16(false, afr, false, bfr,
                                                       (short)0, acc[j], false, false);
    }
  }
  int mbase = tm*16 + ((lane >= 16) ? 8 : 0);
  #pragma unroll
  for (int j = 0; j < 4; j++){
    int ncol = tg*64 + j*16 + lrow;
    float bv = bias ? bias[ncol] : 0.0f;
    #pragma unroll
    for (int v = 0; v < 8; v++){
      float r = acc[j][v] + bv;
      size_t off = (size_t)(mbase + v)*ldc + coloff + ncol;
      if (Cf) Cf[off] = r; else Cbf[off] = (bf16_t)r;
    }
  }
}

// ---------------- relu + layernorm (wave per 512-wide row) -> bf16 ----------------
__global__ void k_relu_ln_bf16(const float* __restrict__ X, const float* __restrict__ g,
                               const float* __restrict__ b, bf16_t* __restrict__ Y,
                               int ld_out, int coloff, int rows){
  int row = blockIdx.x*(blockDim.x >> 5) + (threadIdx.x >> 5);
  if (row >= rows) return;
  int lane = threadIdx.x & 31;
  float v[16], s = 0.f, s2 = 0.f;
  #pragma unroll
  for (int i = 0; i < 16; i++){
    float x = X[(size_t)row*H_ + lane + 32*i];
    x = fmaxf(x, 0.0f);
    v[i] = x; s += x; s2 += x*x;
  }
  s = wredf(s); s2 = wredf(s2);
  float mu = s * (1.0f/H_);
  float var = s2 * (1.0f/H_) - mu*mu;
  float rs = rsqrtf(var + 1e-5f);
  #pragma unroll
  for (int i = 0; i < 16; i++){
    int c = lane + 32*i;
    Y[(size_t)row*ld_out + coloff + c] = (bf16_t)((v[i]-mu)*rs*g[c] + b[c]);
  }
}

// ---------------- one LSTM timestep (sequential dep across launches) ----------------
// gates = gates_x[:,t,:] + h @ W_hh ; fused i/f/g/o tiles per wave.
__global__ void k_lstm_step(const bf16_t* __restrict__ hin, const bf16_t* __restrict__ Whh_t,
                            const float* __restrict__ gx, float* __restrict__ cst,
                            float* __restrict__ hs, bf16_t* __restrict__ hout, int t){
  int wid = blockIdx.x*(blockDim.x >> 5) + (threadIdx.x >> 5);  // 0..127
  int tm = wid >> 5;            // batch tile 0..3
  int n0 = (wid & 31) * 16;     // hidden col 0..496
  int lane = threadIdx.x & 31;
  int lrow = lane & 15;
  int kA = (lane >> 4) * 16;
  int kB = (lane >> 4) * 16;
  const bf16_t* pa = hin + (size_t)(tm*16 + lrow)*H_ + kA;
  v8f acc[4] = {};
  for (int k0 = 0; k0 < H_; k0 += 32){
    v8bf alo = *(const v8bf*)(pa + k0);
    v8bf ahi = *(const v8bf*)(pa + k0 + 8);
    v16bf afr;
    #pragma unroll
    for (int i = 0; i < 8; i++){ afr[i] = alo[i]; afr[i+8] = ahi[i]; }
    #pragma unroll
    for (int gI = 0; gI < 4; gI++){
      const bf16_t* pb = Whh_t + (size_t)(gI*H_ + n0 + lrow)*H_ + kB + k0;
      v16bf bfr = *(const v16bf*)pb;
      acc[gI] = __builtin_amdgcn_wmma_f32_16x16x32_bf16(false, afr, false, bfr,
                                                        (short)0, acc[gI], false, false);
    }
  }
  int n = n0 + lrow;
  int mbase = tm*16 + ((lane >= 16) ? 8 : 0);
  #pragma unroll
  for (int v = 0; v < 8; v++){
    int m = mbase + v;                               // batch index
    size_t gbase = ((size_t)m*S_ + t)*G4_ + n;
    float iv = gx[gbase + 0*H_] + acc[0][v];
    float fv = gx[gbase + 1*H_] + acc[1][v];
    float gv = gx[gbase + 2*H_] + acc[2][v];
    float ov = gx[gbase + 3*H_] + acc[3][v];
    float c  = cst[(size_t)m*H_ + n];
    c = sigmoidf_(fv)*c + sigmoidf_(iv)*tanhf(gv);
    float h = sigmoidf_(ov)*tanhf(c);
    cst[(size_t)m*H_ + n] = c;
    hs[((size_t)m*S_ + t)*H_ + n] = h;
    hout[(size_t)m*H_ + n] = (bf16_t)h;
  }
}

// ---------------- heads: wave per token ----------------
__device__ __forceinline__ void head_eval(const float hreg[16], int lane,
    const float* __restrict__ wt, const float* __restrict__ bias,
    const int* __restrict__ avail, int samp, int NH, float* lp, float* ent){
  float mmax = -INFINITY, ssum = 0.f, tsum = 0.f, lsamp = 0.f;
  for (int n = 0; n < NH; n++){
    float p = 0.f;
    #pragma unroll
    for (int i = 0; i < 16; i++) p += hreg[i]*wt[(size_t)n*H_ + lane + 32*i];
    p = wredf(p);
    float l = p + bias[n] + ((avail[n] == 0) ? -1e16f : 1.0f);
    if (n == samp) lsamp = l;
    if (l > mmax){
      float e = __expf(mmax - l);
      ssum = ssum*e + 1.0f; tsum = tsum*e + l; mmax = l;
    } else {
      float e = __expf(l - mmax);
      ssum += e; tsum += e*l;
    }
  }
  float lse = mmax + __logf(ssum);
  *lp  = lsamp - lse;
  *ent = lse - tsum/ssum;
}

__global__ void k_heads(const float* __restrict__ hs,
    const float* __restrict__ wtv, const float* __restrict__ bv,
    const float* __restrict__ wta, const float* __restrict__ ba,
    const int* __restrict__ av_a, const int* __restrict__ sm_a, const float* __restrict__ on_a,
    const float* __restrict__ wtm, const float* __restrict__ bm,
    const int* __restrict__ av_m, const int* __restrict__ sm_m, const float* __restrict__ on_m,
    const float* __restrict__ wtt, const float* __restrict__ bt,
    const int* __restrict__ av_t, const int* __restrict__ sm_t, const float* __restrict__ on_t,
    float* __restrict__ out){
  int token = blockIdx.x*(blockDim.x >> 5) + (threadIdx.x >> 5);
  if (token >= T_) return;
  int lane = threadIdx.x & 31;
  float hreg[16];
  #pragma unroll
  for (int i = 0; i < 16; i++) hreg[i] = hs[(size_t)token*H_ + lane + 32*i];
  // value head
  for (int n = 0; n < NBINS_; n++){
    float p = 0.f;
    #pragma unroll
    for (int i = 0; i < 16; i++) p += hreg[i]*wtv[(size_t)n*H_ + lane + 32*i];
    p = wredf(p);
    if (lane == 0) out[2*T_ + (size_t)token*NBINS_ + n] = p + bv[n];
  }
  float lpa, ea, lpm, em, lpt, et;
  head_eval(hreg, lane, wta, ba, av_a + (size_t)token*NA_, sm_a[token], NA_, &lpa, &ea);
  head_eval(hreg, lane, wtm, bm, av_m + (size_t)token*NM_, sm_m[token], NM_, &lpm, &em);
  head_eval(hreg, lane, wtt, bt, av_t + (size_t)token*NT_, sm_t[token], NT_, &lpt, &et);
  if (lane == 0){
    out[token]      = on_a[token]*lpa + on_m[token]*lpm + on_t[token]*lpt;
    out[T_ + token] = on_a[token]*ea  + on_m[token]*em  + on_t[token]*et;
  }
}

// ---------------- host ----------------
extern "C" void kernel_launch(void* const* d_in, const int* in_sizes, int n_in,
                              void* d_out, int out_size, void* d_ws, size_t ws_size,
                              hipStream_t stream){
  (void)in_sizes; (void)n_in; (void)out_size; (void)ws_size;
  const float* obs_mine  = (const float*)d_in[0];
  const float* obs_ally  = (const float*)d_in[1];
  const float* obs_enemy = (const float*)d_in[2];
  const int*   avail_act = (const int*)d_in[3];
  const int*   avail_mov = (const int*)d_in[4];
  const int*   avail_tgt = (const int*)d_in[5];
  const float* on_act    = (const float*)d_in[6];
  const float* on_mov    = (const float*)d_in[7];
  const float* on_tgt    = (const float*)d_in[8];
  const int*   smp_act   = (const int*)d_in[9];
  const int*   smp_mov   = (const int*)d_in[10];
  const int*   smp_tgt   = (const int*)d_in[11];
  const float* hx        = (const float*)d_in[12];
  const float* cx        = (const float*)d_in[13];
  const float* W_mine = (const float*)d_in[14]; const float* b_mine = (const float*)d_in[15];
  const float* W_ally = (const float*)d_in[16]; const float* b_ally = (const float*)d_in[17];
  const float* W_enem = (const float*)d_in[18]; const float* b_enem = (const float*)d_in[19];
  const float* W_body = (const float*)d_in[20]; const float* b_body = (const float*)d_in[21];
  const float* W_attn = (const float*)d_in[22]; const float* b_attn = (const float*)d_in[23];
  const float* lnm_g = (const float*)d_in[24];  const float* lnm_b = (const float*)d_in[25];
  const float* lnb_g = (const float*)d_in[26];  const float* lnb_b = (const float*)d_in[27];
  const float* lna_g = (const float*)d_in[28];  const float* lna_b = (const float*)d_in[29];
  const float* W_ih  = (const float*)d_in[30];  const float* b_ih  = (const float*)d_in[31];
  const float* W_hh  = (const float*)d_in[32];  const float* b_hh  = (const float*)d_in[33];
  const float* W_val = (const float*)d_in[34];  const float* b_val = (const float*)d_in[35];
  const float* W_act = (const float*)d_in[36];  const float* b_act = (const float*)d_in[37];
  const float* W_mov = (const float*)d_in[38];  const float* b_mov = (const float*)d_in[39];
  const float* W_tgt = (const float*)d_in[40];  const float* b_tgt = (const float*)d_in[41];
  float* out = (float*)d_out;

  char* wp = (char*)d_ws;
  auto carve = [&](size_t bytes) -> void* {
    void* r = (void*)wp; wp += (bytes + 255) & ~(size_t)255; return r;
  };
  bf16_t* wt_mine = (bf16_t*)carve((size_t)H_*DM_*2);
  bf16_t* wt_ally = (bf16_t*)carve((size_t)H_*DA_*2);
  bf16_t* wt_enem = (bf16_t*)carve((size_t)H_*DE_*2);
  bf16_t* wt_body = (bf16_t*)carve((size_t)H_*3*H_*2);
  bf16_t* wt_attn = (bf16_t*)carve((size_t)H_*H_*2);
  bf16_t* wt_ih   = (bf16_t*)carve((size_t)G4_*H_*2);
  bf16_t* wt_hh   = (bf16_t*)carve((size_t)G4_*H_*2);
  float*  wt_val  = (float*)carve((size_t)NBINS_*H_*4);
  float*  wt_act  = (float*)carve((size_t)NA_*H_*4);
  float*  wt_mov  = (float*)carve((size_t)NM_*H_*4);
  float*  wt_tgt  = (float*)carve((size_t)NT_*H_*4);
  float*  bias_g  = (float*)carve((size_t)G4_*4);
  bf16_t* xm_bf   = (bf16_t*)carve((size_t)T_*DM_*2);
  bf16_t* xa_bf   = (bf16_t*)carve((size_t)T_*DA_*2);
  bf16_t* xe_bf   = (bf16_t*)carve((size_t)T_*DE_*2);
  float*  pre     = (float*)carve((size_t)T_*H_*4);
  bf16_t* cat     = (bf16_t*)carve((size_t)T_*3*H_*2);
  bf16_t* bodyact = (bf16_t*)carve((size_t)T_*H_*2);
  bf16_t* encbf   = (bf16_t*)carve((size_t)T_*H_*2);
  float*  gatesx  = (float*)carve((size_t)T_*G4_*4);
  float*  hsbuf   = (float*)carve((size_t)T_*H_*4);
  bf16_t* hbuf0   = (bf16_t*)carve((size_t)B_*H_*2);
  bf16_t* hbuf1   = (bf16_t*)carve((size_t)B_*H_*2);
  float*  cstate  = (float*)carve((size_t)B_*H_*4);

  auto blk = [](size_t n, int t){ return (unsigned)((n + t - 1)/t); };

  // weight prep (every call: launch must be deterministic, no caching)
  k_transpose_bf16<<<blk((size_t)DM_*H_,256),256,0,stream>>>(W_mine, wt_mine, DM_, H_);
  k_transpose_bf16<<<blk((size_t)DA_*H_,256),256,0,stream>>>(W_ally, wt_ally, DA_, H_);
  k_transpose_bf16<<<blk((size_t)DE_*H_,256),256,0,stream>>>(W_enem, wt_enem, DE_, H_);
  k_transpose_bf16<<<blk((size_t)3*H_*H_,256),256,0,stream>>>(W_body, wt_body, 3*H_, H_);
  k_transpose_bf16<<<blk((size_t)H_*H_,256),256,0,stream>>>(W_attn, wt_attn, H_, H_);
  k_transpose_bf16<<<blk((size_t)H_*G4_,256),256,0,stream>>>(W_ih, wt_ih, H_, G4_);
  k_transpose_bf16<<<blk((size_t)H_*G4_,256),256,0,stream>>>(W_hh, wt_hh, H_, G4_);
  k_transpose_f32<<<blk((size_t)H_*NBINS_,256),256,0,stream>>>(W_val, wt_val, H_, NBINS_);
  k_transpose_f32<<<blk((size_t)H_*NA_,256),256,0,stream>>>(W_act, wt_act, H_, NA_);
  k_transpose_f32<<<blk((size_t)H_*NM_,256),256,0,stream>>>(W_mov, wt_mov, H_, NM_);
  k_transpose_f32<<<blk((size_t)H_*NT_,256),256,0,stream>>>(W_tgt, wt_tgt, H_, NT_);
  k_addbias<<<blk(G4_,256),256,0,stream>>>(b_ih, b_hh, bias_g, G4_);
  k_init_state<<<blk((size_t)B_*H_,256),256,0,stream>>>(hx, cx, hbuf0, cstate, B_*H_);

  // symlog + bf16 cast of observations
  k_symlog_bf16<<<blk((size_t)T_*DM_,256),256,0,stream>>>(obs_mine, xm_bf, T_*DM_);
  k_symlog_bf16<<<blk((size_t)T_*DA_,256),256,0,stream>>>(obs_ally, xa_bf, T_*DA_);
  k_symlog_bf16<<<blk((size_t)T_*DE_,256),256,0,stream>>>(obs_enemy, xe_bf, T_*DE_);

  // waves = (M/16)*(N/64); 8 waves per 256-thread block
  unsigned gemm_blk = (unsigned)(((size_t)(T_/16)*(H_/64) + 7)/8);
  // mine: symlog(x)@W + b -> pre ; then relu+LN -> cat[:, 0:512]
  k_gemm_bf16<<<gemm_blk,256,0,stream>>>(xm_bf, wt_mine, b_mine, pre, nullptr, T_, H_, DM_, H_, 0);
  k_relu_ln_bf16<<<T_/8,256,0,stream>>>(pre, lnm_g, lnm_b, cat, 3*H_, 0, T_);
  // ally / enemy -> straight into cat columns (bias only, no activation)
  k_gemm_bf16<<<gemm_blk,256,0,stream>>>(xa_bf, wt_ally, b_ally, nullptr, cat, T_, H_, DA_, 3*H_, H_);
  k_gemm_bf16<<<gemm_blk,256,0,stream>>>(xe_bf, wt_enem, b_enem, nullptr, cat, T_, H_, DE_, 3*H_, 2*H_);
  // body: cat[T x 1536] @ W_body -> pre ; relu+LN -> bodyact
  k_gemm_bf16<<<gemm_blk,256,0,stream>>>(cat, wt_body, b_body, pre, nullptr, T_, H_, 3*H_, H_, 0);
  k_relu_ln_bf16<<<T_/8,256,0,stream>>>(pre, lnb_g, lnb_b, bodyact, H_, 0, T_);
  // attn: bodyact @ W_attn -> pre ; relu+LN -> encbf
  k_gemm_bf16<<<gemm_blk,256,0,stream>>>(bodyact, wt_attn, b_attn, pre, nullptr, T_, H_, H_, H_, 0);
  k_relu_ln_bf16<<<T_/8,256,0,stream>>>(pre, lna_g, lna_b, encbf, H_, 0, T_);
  // gates_x = enc @ W_ih + (b_ih + b_hh)   [T x 2048]
  unsigned gblk = (unsigned)(((size_t)(T_/16)*(G4_/64) + 7)/8);
  k_gemm_bf16<<<gblk,256,0,stream>>>(encbf, wt_ih, bias_g, gatesx, nullptr, T_, G4_, H_, G4_, 0);

  // sequential LSTM: 512 per-step launches, ping-pong bf16 h
  for (int t = 0; t < S_; t++){
    const bf16_t* hin = (t & 1) ? hbuf1 : hbuf0;
    bf16_t* hout      = (t & 1) ? hbuf0 : hbuf1;
    k_lstm_step<<<16,256,0,stream>>>(hin, wt_hh, gatesx, cstate, hsbuf, hout, t);
  }

  // heads: log_probs | entropy | value
  k_heads<<<T_/8,256,0,stream>>>(hsbuf,
      wt_val, b_val,
      wt_act, b_act, avail_act, smp_act, on_act,
      wt_mov, b_mov, avail_mov, smp_mov, on_mov,
      wt_tgt, b_tgt, avail_tgt, smp_tgt, on_tgt,
      out);
}
